// GraphModel_28965259444614
// MI455X (gfx1250) — matrix operand order repaired and checked by reading, not compiled, not run on hardware
//
#include <hip/hip_runtime.h>
#include <cstdint>
#include <cstddef>

#define F_IN  128
#define HID   128
#define NOUT  64

typedef __attribute__((ext_vector_type(16))) __bf16 v16bf;
typedef __attribute__((ext_vector_type(8)))  float  v8f;

// ---------------------------------------------------------------------------
// Weight prep: W[K][N] fp32 -> Wt[N][K] bf16 (transposed so B-fragment loads
// are contiguous along K).
// ---------------------------------------------------------------------------
__global__ void k_transpose_to_bf16(const float* __restrict__ W,
                                    __bf16* __restrict__ Wt, int K, int N) {
  int idx = blockIdx.x * blockDim.x + threadIdx.x;
  if (idx >= K * N) return;
  int n = idx / K;
  int k = idx - n * K;
  Wt[(size_t)n * K + k] = (__bf16)W[(size_t)k * N + n];
}

// ---------------------------------------------------------------------------
// Degree / symmetric norm. deg starts at 1 (self loop), +1 per incoming edge,
// then in-place rsqrt.
// ---------------------------------------------------------------------------
__global__ void k_deg_init(float* deg, int n) {
  int i = blockIdx.x * blockDim.x + threadIdx.x;
  if (i < n) deg[i] = 1.0f;
}
__global__ void k_deg_accum(const int* __restrict__ dst, float* deg, int E) {
  int i = blockIdx.x * blockDim.x + threadIdx.x;
  if (i < E) atomicAdd(&deg[dst[i]], 1.0f);
}
__global__ void k_deg_to_dinv(float* deg, int n) {
  int i = blockIdx.x * blockDim.x + threadIdx.x;
  if (i < n) deg[i] = rsqrtf(deg[i]);
}

// ---------------------------------------------------------------------------
// agg[i][f] = bias[f] + h[i][f] * dinv[i]^2   (self-loop + bias pre-add)
// ---------------------------------------------------------------------------
__global__ void k_agg_init(const float* __restrict__ h, const float* __restrict__ dinv,
                           const float* __restrict__ bias, float* __restrict__ agg,
                           int n, int F) {
  int idx = blockIdx.x * blockDim.x + threadIdx.x;
  int perRow = F >> 2;
  int i = idx / perRow;
  if (i >= n) return;
  int f = (idx - i * perRow) << 2;
  float w = dinv[i]; w *= w;
  const float4 hv = *(const float4*)(h + (size_t)i * F + f);
  float4 o;
  o.x = bias[f + 0] + hv.x * w;
  o.y = bias[f + 1] + hv.y * w;
  o.z = bias[f + 2] + hv.z * w;
  o.w = bias[f + 3] + hv.w * w;
  *(float4*)(agg + (size_t)i * F + f) = o;
}

// ---------------------------------------------------------------------------
// Edge scatter: one wave per edge (32 lanes x float4 = 128 feats). Gathers are
// coalesced 512B per wave; scatter uses global_atomic_add_f32 (L2-resident).
// ---------------------------------------------------------------------------
__global__ void k_edge_scatter(const float* __restrict__ h, const int* __restrict__ src,
                               const int* __restrict__ dst, const float* __restrict__ dinv,
                               float* __restrict__ agg, int E, int F) {
  int t = blockIdx.x * blockDim.x + threadIdx.x;
  int perEdge = F >> 2;                       // 32 lanes per edge when F==128
  int e = t / perEdge;
  if (e >= E) return;
  int f = (t - e * perEdge) << 2;
  int s = src[e], d = dst[e];
  float w = dinv[s] * dinv[d];
  const float4 hv = *(const float4*)(h + (size_t)s * F + f);
  float* out = agg + (size_t)d * F + f;
  atomicAdd(out + 0, hv.x * w);
  atomicAdd(out + 1, hv.y * w);
  atomicAdd(out + 2, hv.z * w);
  atomicAdd(out + 3, hv.w * w);
}

__global__ void k_tanh_inplace(float* h, int total) {
  int i = blockIdx.x * blockDim.x + threadIdx.x;
  if (i < total) h[i] = tanhf(h[i]);
}

// ---------------------------------------------------------------------------
// Fragment loaders (addresses are base + compile-time immediate offsets).
// ---------------------------------------------------------------------------
__device__ __forceinline__ v16bf load_a_frag(const float* __restrict__ p) {
  v16bf f;
#pragma unroll
  for (int e = 0; e < 8; ++e) {
    f[e]     = (__bf16)p[e];          // fp32 -> bf16 via v_cvt_pk_bf16_f32
    f[e + 8] = (__bf16)p[e + 8];
  }
  return f;
}
__device__ __forceinline__ v16bf load_b_frag(const __bf16* __restrict__ p) {
  v16bf f;
#pragma unroll
  for (int e = 0; e < 16; ++e) f[e] = p[e];
  return f;
}

// ---------------------------------------------------------------------------
// GEMM: C[M x NT] = A[M x K](f32) @ Wt[NT x K](bf16)^T (+bias).
// One wave computes a 16x64 tile (A-fragment reused across 4 column subtiles).
// K/NT are compile-time: the K-loop fully unrolls into 16 straight-line
// v_wmma_f32_16x16x32_bf16. Explicit double-buffering: the NEXT k-step's
// A + 4 B fragments are loaded into fresh registers before the CURRENT step's
// WMMAs issue, so every WMMA consumes operands fetched a full step earlier
// (no load->s_wait_loadcnt 0->wmma serialization in steady state).
// Fragment layouts (C/D per CDNA5 ISA 7.12.2; A/B K order is this code's own):
//   A 16x32 bf16 : lane&15 = M, elem e -> K=half*16+e (contiguous, as B)
//   B 32x16 bf16 : lane&15 = N, K contiguous: half 0 -> K=0..15, half 1 -> 16..31
//   C/D 16x16 f32: lane 0-15 -> M=r, lane 16-31 -> M=8+r, N = lane&15
// M (=50000), NT (128/64), K (128) are exactly tile-divisible: no tails, and
// out-of-range waves return as whole waves so EXEC is all-ones at every WMMA.
// ---------------------------------------------------------------------------
template <int K, int NT, bool HAS_BIAS>
__global__ __launch_bounds__(256)
void k_gemm_bf16(const float* __restrict__ A, const __bf16* __restrict__ Wt,
                 const float* __restrict__ bias, float* __restrict__ C, int M) {
  const int lane = threadIdx.x & 31;
  const int wave = blockIdx.x * (blockDim.x >> 5) + (threadIdx.x >> 5);
  constexpr int colTiles = NT >> 6;           // 64-wide column tiles
  const int rowTile = wave / colTiles;
  const int colTile = wave - rowTile * colTiles;
  if (rowTile * 16 >= M) return;              // wave-uniform exit

  const int mSub = lane & 15;
  const int half = lane >> 4;

  // Loop-invariant base pointers; all loads use immediate offsets from these.
  const float*  abase = A + (size_t)(rowTile * 16 + mSub) * K + half * 16;
  const __bf16* bb[4];
#pragma unroll
  for (int s = 0; s < 4; ++s)
    bb[s] = Wt + (size_t)(colTile * 64 + s * 16 + mSub) * K + half * 16;

  v8f acc[4] = {};

  constexpr int STEPS = K / 32;
  v16bf aCur = load_a_frag(abase);
  v16bf bCur[4];
#pragma unroll
  for (int s = 0; s < 4; ++s) bCur[s] = load_b_frag(bb[s]);

#pragma unroll
  for (int i = 0; i < STEPS; ++i) {
    v16bf aNxt = {};
    v16bf bNxt[4] = {};
    if (i + 1 < STEPS) {                      // compile-time after unroll
      const int kn = (i + 1) * 32;
      aNxt = load_a_frag(abase + kn);
#pragma unroll
      for (int s = 0; s < 4; ++s) bNxt[s] = load_b_frag(bb[s] + kn);
    }
#pragma unroll
    for (int s = 0; s < 4; ++s)
      acc[s] = __builtin_amdgcn_wmma_f32_16x16x32_bf16(
          /*neg_a=*/false, aCur, /*neg_b=*/false, bCur[s],
          /*c_mod=*/(short)0, acc[s], /*reuse_a=*/false, /*reuse_b=*/false);
    aCur = aNxt;
#pragma unroll
    for (int s = 0; s < 4; ++s) bCur[s] = bNxt[s];
  }

  // Branch-free epilogue with immediate store strides.
  const int mBase = rowTile * 16 + half * 8;
#pragma unroll
  for (int s = 0; s < 4; ++s) {
    const int ncol = colTile * 64 + s * 16 + mSub;
    float bv = 0.0f;
    if constexpr (HAS_BIAS) bv = bias[ncol];
    float* cbase = C + (size_t)mBase * NT + ncol;
#pragma unroll
    for (int r = 0; r < 8; ++r)
      cbase[(size_t)r * NT] = acc[s][r] + bv;
  }
}

// ---------------------------------------------------------------------------
// Column-wise max over nodes, two stages (blockDim.x == F == 64).
// ---------------------------------------------------------------------------
__global__ void k_max_partial(const float* __restrict__ h, float* __restrict__ partial,
                              int n, int F) {
  int f = threadIdx.x;
  int chunk = (n + gridDim.x - 1) / gridDim.x;
  int start = blockIdx.x * chunk;
  int end = start + chunk; if (end > n) end = n;
  float m = -__builtin_inff();
  for (int i = start; i < end; ++i)
    m = fmaxf(m, h[(size_t)i * F + f]);
  partial[(size_t)blockIdx.x * F + f] = m;
}
__global__ void k_max_final(const float* __restrict__ partial, float* __restrict__ out,
                            int nb, int F) {
  int f = threadIdx.x;
  float m = -__builtin_inff();
  for (int b = 0; b < nb; ++b)
    m = fmaxf(m, partial[(size_t)b * F + f]);
  out[f] = m;
}

// ---------------------------------------------------------------------------
extern "C" void kernel_launch(void* const* d_in, const int* in_sizes, int n_in,
                              void* d_out, int out_size, void* d_ws, size_t ws_size,
                              hipStream_t stream) {
  const float* x  = (const float*)d_in[0];
  const int*   ei = (const int*)d_in[1];
  const float* W1 = (const float*)d_in[2];
  const float* b1 = (const float*)d_in[3];
  const float* Wl = (const float*)d_in[4];
  const float* bl = (const float*)d_in[5];
  const float* W2 = (const float*)d_in[6];
  const float* b2 = (const float*)d_in[7];
  const float* Wo = (const float*)d_in[8];
  const float* bo = (const float*)d_in[9];
  float* out = (float*)d_out;

  const int N = in_sizes[0] / F_IN;   // 50000
  const int E = in_sizes[1] / 2;      // 800000
  const int* src = ei;
  const int* dst = ei + E;

  // workspace carve-out
  char* ws = (char*)d_ws;
  size_t off = 0;
  auto carve = [&](size_t bytes) -> void* {
    void* p = ws + off;
    off = (off + bytes + 255) & ~(size_t)255;
    return p;
  };
  float*  dinv = (float*)carve((size_t)N * 4);           // deg -> dinv in place
  float*  hA   = (float*)carve((size_t)N * HID * 4);
  float*  hB   = (float*)carve((size_t)N * HID * 4);
  __bf16* Wt1  = (__bf16*)carve((size_t)HID * F_IN * 2);
  __bf16* Wtl  = (__bf16*)carve((size_t)HID * HID * 2);
  __bf16* Wt2  = (__bf16*)carve((size_t)HID * HID * 2);
  __bf16* Wto  = (__bf16*)carve((size_t)NOUT * HID * 2);
  const int MAXBLK = 256;
  float* partial = (float*)carve((size_t)MAXBLK * NOUT * 4);
  (void)ws_size; (void)n_in; (void)out_size;

  // weights -> bf16 transposed [n][k]
  k_transpose_to_bf16<<<(128 * 128 + 255) / 256, 256, 0, stream>>>(W1, Wt1, 128, 128);
  k_transpose_to_bf16<<<(128 * 128 + 255) / 256, 256, 0, stream>>>(Wl, Wtl, 128, 128);
  k_transpose_to_bf16<<<(128 * 128 + 255) / 256, 256, 0, stream>>>(W2, Wt2, 128, 128);
  k_transpose_to_bf16<<<(128 * 64  + 255) / 256, 256, 0, stream>>>(Wo, Wto, 128, 64);

  // degree / dinv
  k_deg_init   <<<(N + 255) / 256, 256, 0, stream>>>(dinv, N);
  k_deg_accum  <<<(E + 255) / 256, 256, 0, stream>>>(dst, dinv, E);
  k_deg_to_dinv<<<(N + 255) / 256, 256, 0, stream>>>(dinv, N);

  auto blocksFor = [&](int Nout) {
    int waves = (N / 16) * (Nout / 64);
    return (waves + 7) / 8;                    // 8 waves per 256-thread block
  };

  const int aggThreads  = N * (HID / 4);
  const int edgeThreads = E * (HID / 4);

  // ---- GCNConv 1: hB = tanh(agg(x @ W1) + b1) ----
  k_gemm_bf16<128, HID, false><<<blocksFor(HID), 256, 0, stream>>>(x, Wt1, nullptr, hA, N);
  k_agg_init    <<<(aggThreads  + 255) / 256, 256, 0, stream>>>(hA, dinv, b1, hB, N, HID);
  k_edge_scatter<<<(edgeThreads + 255) / 256, 256, 0, stream>>>(hA, src, dst, dinv, hB, E, HID);
  k_tanh_inplace<<<(N * HID + 255) / 256, 256, 0, stream>>>(hB, N * HID);

  // ---- linear: hA = hB @ Wl + bl ----
  k_gemm_bf16<128, HID, true><<<blocksFor(HID), 256, 0, stream>>>(hB, Wtl, bl, hA, N);

  // ---- GCNConv 2: hA = tanh(agg(hA @ W2) + b2) ----
  k_gemm_bf16<128, HID, false><<<blocksFor(HID), 256, 0, stream>>>(hA, Wt2, nullptr, hB, N);
  k_agg_init    <<<(aggThreads  + 255) / 256, 256, 0, stream>>>(hB, dinv, b2, hA, N, HID);
  k_edge_scatter<<<(edgeThreads + 255) / 256, 256, 0, stream>>>(hB, src, dst, dinv, hA, E, HID);
  k_tanh_inplace<<<(N * HID + 255) / 256, 256, 0, stream>>>(hA, N * HID);

  // ---- output: hB[:, :64] = hA @ Wo + bo ----
  k_gemm_bf16<128, NOUT, true><<<blocksFor(NOUT), 256, 0, stream>>>(hA, Wto, bo, hB, N);

  // ---- adaptive max pool over nodes ----
  k_max_partial<<<MAXBLK, NOUT, 0, stream>>>(hB, partial, N, NOUT);
  k_max_final  <<<1, NOUT, 0, stream>>>(partial, out, MAXBLK, NOUT);
}
